// GINClassifier_73229192397464
// MI455X (gfx1250) — compile-verified, matrix operand order checked by reading, not run on hardware
//
#include <hip/hip_runtime.h>

#define NODES 100000
#define EDGES 800000
#define GRAPHS 128

typedef __attribute__((ext_vector_type(16))) __bf16 v16bf;
typedef __attribute__((ext_vector_type(8)))  __bf16 v8bf;
typedef __attribute__((ext_vector_type(4)))  __bf16 v4bf;
typedef __attribute__((ext_vector_type(8)))  float  v8f;

// Build a 16-element bf16 WMMA fragment from two contiguous 8xbf16 (16B) LDS runs.
// Per CDNA5 16-bit A/B layout: lane-half 0 holds K = {0..7, 16..23}, half 1 holds
// K = {8..15, 24..31}; both runs are contiguous in K -> two ds_load_b128.
__device__ __forceinline__ v16bf make_frag(const __bf16* p) {
    v8bf lo = *(const v8bf*)p;
    v8bf hi = *(const v8bf*)(p + 16);
    v16bf f;
#pragma unroll
    for (int i = 0; i < 8; ++i) { f[i] = lo[i]; f[i + 8] = hi[i]; }
    return f;
}

// agg = (1 + eps) * x   (fused GIN self-term), float4-vectorized
__global__ void k_init_agg(const float4* __restrict__ x, const float* __restrict__ eps,
                           float4* __restrict__ agg, long long total4) {
    long long i = (long long)blockIdx.x * blockDim.x + threadIdx.x;
    if (i < total4) {
        float s = 1.0f + eps[0];
        float4 v = x[i];
        v.x *= s; v.y *= s; v.z *= s; v.w *= s;
        agg[i] = v;
    }
}

// Edge-parallel scatter-add: agg[dst] += x[src]. Consecutive threads handle
// consecutive float4 chunks of one edge -> coalesced row reads; the agg buffer
// (<=153.6MB) is L2-resident on MI455X (192MB L2) so atomics resolve on-chip.
template <int F4>
__global__ void k_scatter(const float* __restrict__ x, const int* __restrict__ src,
                          const int* __restrict__ dst, float* __restrict__ agg,
                          long long total) {
    long long t = (long long)blockIdx.x * blockDim.x + threadIdx.x;
    if (t >= total) return;
    int e = (int)(t / F4);      // constant divisor -> mul/shift
    int c = (int)(t % F4);
    int s = src[e], d = dst[e];
    const float4 val = ((const float4*)(x + (size_t)s * (F4 * 4)))[c];
    float* p = agg + (size_t)d * (F4 * 4) + c * 4;
    atomicAdd(p + 0, val.x);
    atomicAdd(p + 1, val.y);
    atomicAdd(p + 2, val.z);
    atomicAdd(p + 3, val.w);
}

// Transpose + convert weight [K][N] f32 -> [N][K] bf16 (column-major for B frags).
__global__ void k_convert_wt(const float* __restrict__ w, __bf16* __restrict__ wt,
                             int K, int N) {
    int i = blockIdx.x * blockDim.x + threadIdx.x;
    if (i < K * N) {
        int k = i / N, n = i % N;
        wt[(size_t)n * K + k] = (__bf16)w[i];
    }
}

// Tall-skinny GEMM: out[M,N] = epilogue(A[M,K] @ W[K,N]).
// Block = 256 threads (8 wave32) -> 128 rows x N cols; double-buffered LDS with
// software pipelining: next tile's global_load_b128s are issued before the WMMA
// chain on the current tile, stored to the alternate buffer after, one barrier/step.
// BN=true : out = relu((A@W)*s + b), s = g*rsqrt(v+eps), b = (ba-m)*s + be
// BN=false: out = relu(A@W + ba)
template <int N, bool BN>
__global__ __launch_bounds__(256) void k_gemm(
    const float* __restrict__ A, const __bf16* __restrict__ Wt,
    const float* __restrict__ lin_b,
    const float* __restrict__ bn_g, const float* __restrict__ bn_b,
    const float* __restrict__ bn_m, const float* __restrict__ bn_v,
    float* __restrict__ out, int M, int K) {
    constexpr int NT = N / 16;   // 16-wide column tiles
    constexpr int WI = N / 64;   // 16B chunks per thread for the W tile
    __shared__ __align__(16) __bf16 ldsA[2][128 * 32];
    __shared__ __align__(16) __bf16 ldsW[2][N * 32];

    const int t = threadIdx.x;
    const int wv = t >> 5;
    const int lane = t & 31;
    const int kh = lane >> 4;   // K-half of this lane (WMMA 16-bit A/B layout)
    const int lr = lane & 15;   // row (A) / col (B) within the 16-wide tile
    const int rowBase = blockIdx.x * 128;
    const int nsteps = K >> 5;

    // --- staging helpers (fixed per-thread coordinates) ---
    auto load_tileA = [&](int k0, float4* regs) {
#pragma unroll
        for (int i = 0; i < 4; ++i) {
            int idx = i * 256 + t;         // 1024 float4 chunks: 128 rows x 8
            int row = idx >> 3, c4 = idx & 7;
            int gr = rowBase + row;
            float4 v = make_float4(0.f, 0.f, 0.f, 0.f);
            if (gr < M) v = *(const float4*)(A + (size_t)gr * K + k0 + c4 * 4);
            regs[i] = v;
        }
    };
    auto store_tileA = [&](int buf, const float4* regs) {
#pragma unroll
        for (int i = 0; i < 4; ++i) {
            int idx = i * 256 + t;
            int row = idx >> 3, c4 = idx & 7;
            v4bf pk;
            pk[0] = (__bf16)regs[i].x; pk[1] = (__bf16)regs[i].y;
            pk[2] = (__bf16)regs[i].z; pk[3] = (__bf16)regs[i].w;
            *(v4bf*)&ldsA[buf][row * 32 + c4 * 4] = pk;
        }
    };
    auto load_tileW = [&](int k0, v8bf* regs) {
#pragma unroll
        for (int i = 0; i < WI; ++i) {
            int idx = i * 256 + t;         // N*4 16B chunks: N cols x 4
            int col = idx >> 2, kc = idx & 3;
            regs[i] = *(const v8bf*)(Wt + (size_t)col * K + k0 + kc * 8);
        }
    };
    auto store_tileW = [&](int buf, const v8bf* regs) {
#pragma unroll
        for (int i = 0; i < WI; ++i) {
            int idx = i * 256 + t;
            int col = idx >> 2, kc = idx & 3;
            *(v8bf*)&ldsW[buf][col * 32 + kc * 8] = regs[i];
        }
    };

    v8f acc[NT];
#pragma unroll
    for (int nt = 0; nt < NT; ++nt)
#pragma unroll
        for (int r = 0; r < 8; ++r) acc[nt][r] = 0.0f;

    // Prologue: fill buffer 0
    float4 aR[4];
    v8bf   wR[WI];
    load_tileA(0, aR);
    load_tileW(0, wR);
    store_tileA(0, aR);
    store_tileW(0, wR);
    __syncthreads();

    for (int ks = 0; ks < nsteps; ++ks) {
        const int cur = ks & 1, nxt = cur ^ 1;
        const bool more = (ks + 1) < nsteps;

        // Issue next tile's global loads early; they overlap the WMMA chain.
        if (more) {
            load_tileA((ks + 1) * 32, aR);
            load_tileW((ks + 1) * 32, wR);
        }

        v16bf afrag = make_frag(&ldsA[cur][(wv * 16 + lr) * 32 + kh * 8]);
#pragma unroll
        for (int nt = 0; nt < NT; ++nt) {
            v16bf bfrag = make_frag(&ldsW[cur][(nt * 16 + lr) * 32 + kh * 8]);
            acc[nt] = __builtin_amdgcn_wmma_f32_16x16x32_bf16(
                false, afrag, false, bfrag, (short)0, acc[nt], false, false);
        }

        if (more) {
            store_tileA(nxt, aR);
            store_tileW(nxt, wR);
        }
        __syncthreads();
    }

    // Epilogue using C/D layout: lane&15 -> N, vgpr r -> M = half*8 + r.
#pragma unroll
    for (int nt = 0; nt < NT; ++nt) {
        int col = nt * 16 + lr;
        float s, b;
        if (BN) {
            s = bn_g[col] * rsqrtf(bn_v[col] + 1e-5f);
            b = (lin_b[col] - bn_m[col]) * s + bn_b[col];
        } else {
            s = 1.0f;
            b = lin_b[col];
        }
#pragma unroll
        for (int r = 0; r < 8; ++r) {
            int row = rowBase + wv * 16 + kh * 8 + r;
            if (row < M) out[(size_t)row * N + col] = fmaxf(acc[nt][r] * s + b, 0.0f);
        }
    }
}

__global__ void k_zero(float* __restrict__ p, int n) {
    int i = blockIdx.x * blockDim.x + threadIdx.x;
    if (i < n) p[i] = 0.0f;
}

// Global add-pool: pooled[batch[node]] += h[node]  (h is [NODES,64])
__global__ void k_pool(const float* __restrict__ h, const int* __restrict__ batch,
                       float* __restrict__ pooled, long long total) {
    long long t = (long long)blockIdx.x * blockDim.x + threadIdx.x;
    if (t >= total) return;
    int node = (int)(t >> 4);
    int c = (int)(t & 15);
    int g = batch[node];
    const float4 v = ((const float4*)(h + (size_t)node * 64))[c];
    float* p = pooled + (size_t)g * 64 + c * 4;
    atomicAdd(p + 0, v.x);
    atomicAdd(p + 1, v.y);
    atomicAdd(p + 2, v.z);
    atomicAdd(p + 3, v.w);
}

__global__ void k_classify(const float* __restrict__ pooled, const float* __restrict__ wc,
                           const float* __restrict__ bc, float* __restrict__ out) {
    int t = threadIdx.x;
    if (t < GRAPHS * 2) {
        int g = t >> 1, c = t & 1;
        float acc = bc[c];
#pragma unroll
        for (int k = 0; k < 64; ++k) acc += pooled[g * 64 + k] * wc[k * 2 + c];
        out[g * 2 + c] = acc;
    }
}

static inline unsigned nblk(long long n, int bt) { return (unsigned)((n + bt - 1) / bt); }

extern "C" void kernel_launch(void* const* d_in, const int* in_sizes, int n_in,
                              void* d_out, int out_size, void* d_ws, size_t ws_size,
                              hipStream_t stream) {
    const float* x     = (const float*)d_in[0];
    const int*   ei    = (const int*)d_in[1];
    const int*   batch = (const int*)d_in[2];
    const float* eps1 = (const float*)d_in[3];
    const float* w1a  = (const float*)d_in[4];
    const float* b1a  = (const float*)d_in[5];
    const float* g1   = (const float*)d_in[6];
    const float* be1  = (const float*)d_in[7];
    const float* m1   = (const float*)d_in[8];
    const float* v1   = (const float*)d_in[9];
    const float* w1b  = (const float*)d_in[10];
    const float* b1b  = (const float*)d_in[11];
    const float* eps2 = (const float*)d_in[12];
    const float* w2a  = (const float*)d_in[13];
    const float* b2a  = (const float*)d_in[14];
    const float* g2   = (const float*)d_in[15];
    const float* be2  = (const float*)d_in[16];
    const float* m2   = (const float*)d_in[17];
    const float* v2   = (const float*)d_in[18];
    const float* w2b  = (const float*)d_in[19];
    const float* b2b  = (const float*)d_in[20];
    const float* eps3 = (const float*)d_in[21];
    const float* w3a  = (const float*)d_in[22];
    const float* b3a  = (const float*)d_in[23];
    const float* g3   = (const float*)d_in[24];
    const float* be3  = (const float*)d_in[25];
    const float* m3   = (const float*)d_in[26];
    const float* v3   = (const float*)d_in[27];
    const float* w3b  = (const float*)d_in[28];
    const float* b3b  = (const float*)d_in[29];
    const float* wc   = (const float*)d_in[30];
    const float* bc   = (const float*)d_in[31];

    const int* src = ei;             // edge_index[0]
    const int* dst = ei + EDGES;     // edge_index[1]

    // Workspace layout
    char* ws = (char*)d_ws;
    size_t offA = 0;                                            // [NODES*384] f32 agg
    size_t offB = offA + (size_t)NODES * 384 * sizeof(float);   // [NODES*128] f32
    size_t offC = offB + (size_t)NODES * 128 * sizeof(float);   // [NODES*128] f32
    size_t offW = offC + (size_t)NODES * 128 * sizeof(float);   // [384*128] bf16 scratch
    size_t offP = offW + ((size_t)384 * 128 * 2 + 255) / 256 * 256;  // [128*64] f32
    float*  bufA   = (float*)(ws + offA);
    float*  bufB   = (float*)(ws + offB);
    float*  bufC   = (float*)(ws + offC);
    __bf16* bufW   = (__bf16*)(ws + offW);
    float*  pooled = (float*)(ws + offP);

    const int BT = 256;
    const unsigned GB = (NODES + 127) / 128;  // GEMM blocks: 782

    // ---- Layer 1 (384 -> 128 -> 128) ----
    k_init_agg<<<nblk((long long)NODES * 96, BT), BT, 0, stream>>>(
        (const float4*)x, eps1, (float4*)bufA, (long long)NODES * 96);
    k_scatter<96><<<nblk((long long)EDGES * 96, BT), BT, 0, stream>>>(x, src, dst, bufA, (long long)EDGES * 96);
    k_convert_wt<<<nblk(384 * 128, BT), BT, 0, stream>>>(w1a, bufW, 384, 128);
    k_gemm<128, true><<<GB, 256, 0, stream>>>(bufA, bufW, b1a, g1, be1, m1, v1, bufB, NODES, 384);
    k_convert_wt<<<nblk(128 * 128, BT), BT, 0, stream>>>(w1b, bufW, 128, 128);
    k_gemm<128, false><<<GB, 256, 0, stream>>>(bufB, bufW, b1b, nullptr, nullptr, nullptr, nullptr, bufC, NODES, 128);

    // ---- Layer 2 (128 -> 128 -> 128) ----
    k_init_agg<<<nblk((long long)NODES * 32, BT), BT, 0, stream>>>(
        (const float4*)bufC, eps2, (float4*)bufA, (long long)NODES * 32);
    k_scatter<32><<<nblk((long long)EDGES * 32, BT), BT, 0, stream>>>(bufC, src, dst, bufA, (long long)EDGES * 32);
    k_convert_wt<<<nblk(128 * 128, BT), BT, 0, stream>>>(w2a, bufW, 128, 128);
    k_gemm<128, true><<<GB, 256, 0, stream>>>(bufA, bufW, b2a, g2, be2, m2, v2, bufB, NODES, 128);
    k_convert_wt<<<nblk(128 * 128, BT), BT, 0, stream>>>(w2b, bufW, 128, 128);
    k_gemm<128, false><<<GB, 256, 0, stream>>>(bufB, bufW, b2b, nullptr, nullptr, nullptr, nullptr, bufC, NODES, 128);

    // ---- Layer 3 (128 -> 128 -> 64) ----
    k_init_agg<<<nblk((long long)NODES * 32, BT), BT, 0, stream>>>(
        (const float4*)bufC, eps3, (float4*)bufA, (long long)NODES * 32);
    k_scatter<32><<<nblk((long long)EDGES * 32, BT), BT, 0, stream>>>(bufC, src, dst, bufA, (long long)EDGES * 32);
    k_convert_wt<<<nblk(128 * 128, BT), BT, 0, stream>>>(w3a, bufW, 128, 128);
    k_gemm<128, true><<<GB, 256, 0, stream>>>(bufA, bufW, b3a, g3, be3, m3, v3, bufB, NODES, 128);
    k_convert_wt<<<nblk(128 * 64, BT), BT, 0, stream>>>(w3b, bufW, 128, 64);
    k_gemm<64, false><<<GB, 256, 0, stream>>>(bufB, bufW, b3b, nullptr, nullptr, nullptr, nullptr, bufC, NODES, 128);

    // ---- Pool + classify ----
    k_zero<<<nblk(GRAPHS * 64, BT), BT, 0, stream>>>(pooled, GRAPHS * 64);
    k_pool<<<nblk((long long)NODES * 16, BT), BT, 0, stream>>>(bufC, batch, pooled, (long long)NODES * 16);
    k_classify<<<1, 256, 0, stream>>>(pooled, wc, bc, (float*)d_out);
}
